// GPT_11785390260961
// MI455X (gfx1250) — matrix operand order repaired and checked by reading, not compiled, not run on hardware
//
#include <hip/hip_runtime.h>

// ---------------- model constants ----------------
#define DEPTH 2
#define L_    2048
#define D_    1024
#define HQ_   16
#define HKV_  4
#define HD_   64
#define F_    1024
#define E_    8
#define V_    32000
#define PROJ_ ((HQ_ + 2*HKV_) * HD_)   // 1536
#define LIMIT_ 7.0f
#define ALPHA_ 1.702f
#define EPS_   1e-5f

// ---------------- CDNA5 WMMA types (gfx1250, wave32) ----------------
typedef __attribute__((ext_vector_type(16))) __bf16        v16bf;
typedef __attribute__((ext_vector_type(8)))  float         v8f;
typedef __attribute__((ext_vector_type(8)))  unsigned int  v8u;
typedef int v4i_ __attribute__((__vector_size__(16)));      // matches builtin proto

#define ASG __attribute__((address_space(1)))
#define ASL __attribute__((address_space(3)))

#if defined(__has_builtin)
#  if __has_builtin(__builtin_amdgcn_global_load_async_to_lds_b128) && \
      __has_builtin(__builtin_amdgcn_s_wait_asynccnt)
#    define HAVE_ASYNC 1
#  endif
#endif
#ifndef HAVE_ASYNC
#  define HAVE_ASYNC 0
#endif

__device__ __forceinline__ unsigned short f2bf(float f) {
    unsigned int u = __builtin_bit_cast(unsigned int, f);
    u += 0x7FFFu + ((u >> 16) & 1u);          // round-to-nearest-even
    return (unsigned short)(u >> 16);
}
__device__ __forceinline__ unsigned int pack2(float lo, float hi) {
    return (unsigned int)f2bf(lo) | ((unsigned int)f2bf(hi) << 16);
}

// =====================================================================
// bf16 WMMA GEMM:  C[M,N] (f32, opt +=) = A[M,K](bf16) * Bt[N,K](bf16)^T
// Block 256 threads = 8 waves; macro-tile 128(M) x 128(N); KT = 32.
// Wave (wm,wn) owns a 32x64 C sub-tile: 2 A frags x 4 B frags ->
// 8 accumulators, 8 WMMAs per K-step. Double-buffered LDS with
// GLOBAL_LOAD_ASYNC_TO_LDS_B128: tile k+1 is in flight (ASYNCcnt<=4
// keeps it outstanding) while tile k runs through the WMMA pipe.
// 80B row stride -> 16B aligned, bank-conflict-free ds_load_b128.
// =====================================================================
template <bool ACCUM>
__global__ void k_gemm(const unsigned short* __restrict__ A,
                       const unsigned short* __restrict__ Bt,
                       float* __restrict__ C, int M, int N, int K)
{
    __shared__ __align__(16) unsigned int Au[2][128][20];
    __shared__ __align__(16) unsigned int Bu[2][128][20];

    const int tid  = threadIdx.x;
    const int lane = tid & 31;
    const int w    = tid >> 5;
    const int wm   = w & 3;          // 0..3 : 32-row M sub-tile
    const int wn   = w >> 2;         // 0..1 : 64-col N sub-tile
    const int lh   = lane >> 4;
    const int ll   = lane & 15;

    const int m0 = blockIdx.x * 128;
    const int n0 = blockIdx.y * 128;

    const int sr = tid >> 2;         // staging row 0..63
    const int sq = tid & 3;          // 16B chunk within a 64B row

    const int ar0 = wm * 32 + ll;
    const int ar1 = ar0 + 16;
    const int bnb = wn * 64 + ll;

    v8f acc[2][4];
    #pragma unroll
    for (int i = 0; i < 2; ++i)
        #pragma unroll
        for (int j = 0; j < 4; ++j) acc[i][j] = {};

    auto issue = [&](int k0, int buf) {
        const unsigned short* a0p = &A[(size_t)(m0 + sr)      * K + k0 + sq * 8];
        const unsigned short* a1p = &A[(size_t)(m0 + sr + 64) * K + k0 + sq * 8];
        const unsigned short* b0p = &Bt[(size_t)(n0 + sr)      * K + k0 + sq * 8];
        const unsigned short* b1p = &Bt[(size_t)(n0 + sr + 64) * K + k0 + sq * 8];
#if HAVE_ASYNC
        __builtin_amdgcn_global_load_async_to_lds_b128(
            (ASG v4i_*)(void*)a0p, (ASL v4i_*)(void*)&Au[buf][sr][sq * 4], 0, 0);
        __builtin_amdgcn_global_load_async_to_lds_b128(
            (ASG v4i_*)(void*)a1p, (ASL v4i_*)(void*)&Au[buf][sr + 64][sq * 4], 0, 0);
        __builtin_amdgcn_global_load_async_to_lds_b128(
            (ASG v4i_*)(void*)b0p, (ASL v4i_*)(void*)&Bu[buf][sr][sq * 4], 0, 0);
        __builtin_amdgcn_global_load_async_to_lds_b128(
            (ASG v4i_*)(void*)b1p, (ASL v4i_*)(void*)&Bu[buf][sr + 64][sq * 4], 0, 0);
#else
        *(uint4*)&Au[buf][sr][sq * 4]      = *(const uint4*)a0p;
        *(uint4*)&Au[buf][sr + 64][sq * 4] = *(const uint4*)a1p;
        *(uint4*)&Bu[buf][sr][sq * 4]      = *(const uint4*)b0p;
        *(uint4*)&Bu[buf][sr + 64][sq * 4] = *(const uint4*)b1p;
#endif
    };

    issue(0, 0);                       // prologue: tile 0 in flight
    int cur = 0;

    for (int k0 = 0; k0 < K; k0 += 32) {
#if HAVE_ASYNC
        if (k0 + 32 < K) {
            issue(k0 + 32, cur ^ 1);                 // tile k+1 in flight
            __builtin_amdgcn_s_wait_asynccnt(4);     // tile k landed; k+1 flies
        } else {
            __builtin_amdgcn_s_wait_asynccnt(0);
        }
#else
        if (k0 + 32 < K) issue(k0 + 32, cur ^ 1);
#endif
        __syncthreads();   // tile k visible to all waves

        v8u afr[2], bfr[4];
        #pragma unroll
        for (int v = 0; v < 8; ++v) {
            int ai = lh * 8 + v;
            afr[0][v] = Au[cur][ar0][ai];
            afr[1][v] = Au[cur][ar1][ai];
            int bi = lh * 8 + v;
            #pragma unroll
            for (int j = 0; j < 4; ++j)
                bfr[j][v] = Bu[cur][bnb + j * 16][bi];
        }
        v16bf Af[2], Bf[4];
        #pragma unroll
        for (int i = 0; i < 2; ++i) Af[i] = __builtin_bit_cast(v16bf, afr[i]);
        #pragma unroll
        for (int j = 0; j < 4; ++j) Bf[j] = __builtin_bit_cast(v16bf, bfr[j]);

        #pragma unroll
        for (int i = 0; i < 2; ++i)
            #pragma unroll
            for (int j = 0; j < 4; ++j)
                acc[i][j] = __builtin_amdgcn_wmma_f32_16x16x32_bf16(
                    false, Af[i], false, Bf[j], (short)0, acc[i][j], false, false);

        __syncthreads();   // tile k consumed; its buffer may be re-filled
        cur ^= 1;
    }

    // writeback: C/D f32 layout (VGPR r -> M = r + 8*lh)
    const int colb = n0 + wn * 64 + ll;
    const int rowb = m0 + wm * 32 + (lh ? 8 : 0);
    #pragma unroll
    for (int r = 0; r < 8; ++r) {
        #pragma unroll
        for (int i = 0; i < 2; ++i) {
            float* p = &C[(size_t)(rowb + i * 16 + r) * N + colb];
            #pragma unroll
            for (int j = 0; j < 4; ++j) {
                if (ACCUM) p[j * 16] += acc[i][j][r];
                else       p[j * 16]  = acc[i][j][r];
            }
        }
    }
}

// =====================================================================
// f32 -> bf16 converters
// =====================================================================
__global__ void k_cvt(const float* __restrict__ s, unsigned short* __restrict__ d, int n)
{
    int i = (blockIdx.x * blockDim.x + threadIdx.x) << 2;
    if (i >= n) return;
    float4 f = *(const float4*)&s[i];
    uint2 p;
    p.x = pack2(f.x, f.y);
    p.y = pack2(f.z, f.w);
    *(uint2*)&d[i] = p;
}

// [K,N] f32 -> [N,K] bf16 (for attn_in_w / attn_out_w), 32x32 LDS tiles
__global__ void k_cvt_t(const float* __restrict__ s, unsigned short* __restrict__ d,
                        int Kd, int Nd)
{
    __shared__ float t[32][33];
    int k0 = blockIdx.x * 32, n0 = blockIdx.y * 32;
    #pragma unroll
    for (int j = 0; j < 4; ++j) {
        int idx = threadIdx.x + j * 256;
        int kk = idx >> 5, nn = idx & 31;
        t[kk][nn] = s[(size_t)(k0 + kk) * Nd + n0 + nn];
    }
    __syncthreads();
    #pragma unroll
    for (int j = 0; j < 4; ++j) {
        int idx = threadIdx.x + j * 256;
        int nn = idx >> 5, kk = idx & 31;
        d[(size_t)(n0 + nn) * Kd + k0 + kk] = f2bf(t[kk][nn]);
    }
}

// =====================================================================
// Elementwise / small kernels
// =====================================================================
__global__ void k_embed(const int* __restrict__ x, const float* __restrict__ emb,
                        float* __restrict__ h)
{
    int t = blockIdx.x;
    const float* src = emb + (size_t)x[t] * D_;
    float* dst = h + (size_t)t * D_;
    for (int d = threadIdx.x; d < D_; d += 256) dst[d] = src[d];
}

__global__ void k_rmsnorm(const float* __restrict__ x, const float* __restrict__ scale,
                          float* __restrict__ y, unsigned short* __restrict__ ybf)
{
    int t = blockIdx.x;
    const float* xr = x + (size_t)t * D_;
    float ss = 0.f;
    for (int k = threadIdx.x; k < D_; k += 256) { float v = xr[k]; ss += v * v; }
    #pragma unroll
    for (int off = 16; off; off >>= 1) ss += __shfl_xor(ss, off);
    __shared__ float wsum[8];
    if ((threadIdx.x & 31) == 0) wsum[threadIdx.x >> 5] = ss;
    __syncthreads();
    float tot = 0.f;
    #pragma unroll
    for (int i = 0; i < 8; ++i) tot += wsum[i];
    float r = rsqrtf(tot / (float)D_ + EPS_);
    for (int k = threadIdx.x; k < D_; k += 256) {
        float v = xr[k] * r * scale[k];
        y[(size_t)t * D_ + k]   = v;
        ybf[(size_t)t * D_ + k] = f2bf(v);
    }
}

// In-place RoPE on q (heads 0..15) and k (heads 16..19); 1 wave/(token,head)
__global__ void k_rope(float* __restrict__ proj)
{
    int wid  = (blockIdx.x * blockDim.x + threadIdx.x) >> 5;
    int lane = threadIdx.x & 31;
    int t = wid / (HQ_ + HKV_);
    int h = wid % (HQ_ + HKV_);
    float* base = proj + (size_t)t * PROJ_ +
                  (h < HQ_ ? h * HD_ : HQ_ * HD_ + (h - HQ_) * HD_);
    float freq  = (lane < 16) ? __powf(1.0f / 1024.0f, (float)lane / 15.0f) : 0.0f;
    float theta = (float)t * freq;
    float c = __cosf(theta), s = __sinf(theta);
    float x1 = base[lane], x2 = base[lane + 32];
    base[lane]      =  x1 * c + x2 * s;
    base[lane + 32] = -x1 * s + x2 * c;
}

// Streaming-softmax attention with sink; reference masks k<=q, so only
// k>q attend. One wave per (q-head, query); emits bf16 for the out-proj.
__global__ void k_attn(const float* __restrict__ proj,
                       const float* __restrict__ sink,
                       unsigned short* __restrict__ obf)
{
    int wid  = (blockIdx.x * blockDim.x + threadIdx.x) >> 5;
    int lane = threadIdx.x & 31;
    int hq = wid >> 11;
    int q  = wid & (L_ - 1);
    int kvh = hq >> 2;
    const int koff = HQ_ * HD_ + kvh * HD_;
    const int voff = (HQ_ + HKV_) * HD_ + kvh * HD_;

    const float* qp = proj + (size_t)q * PROJ_ + hq * HD_;
    float q0 = qp[lane], q1 = qp[lane + 32];

    float m = sink[hq];
    float s = 1.0f;              // exp(sink - m)
    float o0 = 0.f, o1 = 0.f;

    for (int k = q + 1; k < L_; ++k) {
        const float* kp = proj + (size_t)k * PROJ_ + koff;
        float p = q0 * kp[lane] + q1 * kp[lane + 32];
        #pragma unroll
        for (int off = 16; off; off >>= 1) p += __shfl_xor(p, off);
        float logit = p * 0.125f;
        float nm = fmaxf(m, logit);
        float sc = __expf(m - nm);
        float pe = __expf(logit - nm);
        const float* vp = proj + (size_t)k * PROJ_ + voff;
        s  = s  * sc + pe;
        o0 = o0 * sc + pe * vp[lane];
        o1 = o1 * sc + pe * vp[lane + 32];
        m = nm;
    }
    float inv = 1.0f / s;
    unsigned short* op = obf + (size_t)q * (HQ_ * HD_) + hq * HD_;
    op[lane]      = f2bf(o0 * inv);
    op[lane + 32] = f2bf(o1 * inv);
}

// Router: gate = u @ gate_w (D x 8), top-2 + softmax. One wave per token.
__global__ void k_gate(const float* __restrict__ u, const float* __restrict__ gw,
                       float* __restrict__ gwts, int* __restrict__ gidx)
{
    int lane = threadIdx.x & 31;
    int t = blockIdx.x * 8 + (threadIdx.x >> 5);
    float a[E_];
    #pragma unroll
    for (int e = 0; e < E_; ++e) a[e] = 0.f;
    for (int k = lane; k < D_; k += 32) {
        float uv = u[(size_t)t * D_ + k];
        #pragma unroll
        for (int e = 0; e < E_; ++e) a[e] += uv * gw[k * E_ + e];
    }
    #pragma unroll
    for (int e = 0; e < E_; ++e) {
        #pragma unroll
        for (int off = 16; off; off >>= 1) a[e] += __shfl_xor(a[e], off);
    }
    if (lane == 0) {
        int i0 = 0;
        #pragma unroll
        for (int e = 1; e < E_; ++e) if (a[e] > a[i0]) i0 = e;
        int i1 = (i0 == 0) ? 1 : 0;
        #pragma unroll
        for (int e = 0; e < E_; ++e) if (e != i0 && a[e] > a[i1]) i1 = e;
        float z  = __expf(a[i1] - a[i0]);
        float p0 = 1.0f / (1.0f + z);
        gwts[t * 2]     = p0;
        gwts[t * 2 + 1] = 1.0f - p0;
        gidx[t * 2]     = i0;
        gidx[t * 2 + 1] = i1;
    }
}

// SwiGLU over interleaved (gate, value) pairs; emits bf16 for the d2 GEMM.
__global__ void k_swiglu(const float* __restrict__ h1, const float* __restrict__ b1,
                         unsigned short* __restrict__ s1bf)
{
    int t = blockIdx.x;
    for (int f = threadIdx.x; f < F_; f += 256) {
        float g  = h1[(size_t)t * 2 * F_ + 2 * f]     + b1[2 * f];
        float xx = h1[(size_t)t * 2 * F_ + 2 * f + 1] + b1[2 * f + 1];
        g  = fminf(g, LIMIT_);
        xx = fminf(fmaxf(xx, -LIMIT_), LIMIT_);
        float sig = 1.0f / (1.0f + __expf(-ALPHA_ * g));
        s1bf[(size_t)t * F_ + f] = f2bf(g * sig * (xx + 1.0f));
    }
}

// h[t,:] += w_e(t) * (h2[t,:] + d2b_e) for tokens routing expert e.
__global__ void k_moe_accum(float* __restrict__ h, const float* __restrict__ h2,
                            const float* __restrict__ b2,
                            const float* __restrict__ gwts,
                            const int* __restrict__ gidx, int e)
{
    int t = blockIdx.x;
    float w = 0.f;
    if (gidx[t * 2]     == e) w += gwts[t * 2];
    if (gidx[t * 2 + 1] == e) w += gwts[t * 2 + 1];
    if (w == 0.f) return;
    for (int d = threadIdx.x; d < D_; d += 256)
        h[(size_t)t * D_ + d] += w * (h2[(size_t)t * D_ + d] + b2[d]);
}

// =====================================================================
// Orchestration (graph-capture safe: stream launches only)
// =====================================================================
extern "C" void kernel_launch(void* const* d_in, const int* in_sizes, int n_in,
                              void* d_out, int out_size, void* d_ws, size_t ws_size,
                              hipStream_t stream)
{
    (void)in_sizes; (void)n_in; (void)out_size; (void)ws_size;

    const int*   x          = (const int*)  d_in[0];
    const float* emb        = (const float*)d_in[1];
    const float* attn_in_w  = (const float*)d_in[2];   // [DEPTH, D, 1536]
    const float* attn_out_w = (const float*)d_in[3];   // [DEPTH, 1024, D]
    const float* sink       = (const float*)d_in[4];   // [DEPTH, 1, HQ]
    const float* ln1        = (const float*)d_in[5];
    const float* ln2        = (const float*)d_in[6];
    const float* lnf        = (const float*)d_in[7];
    const float* gate_w     = (const float*)d_in[8];   // [DEPTH, D, E]
    const float* d1w        = (const float*)d_in[9];   // [DEPTH, E, 2F, D]
    const float* d1b        = (const float*)d_in[10];  // [DEPTH, E, 2F]
    const float* d2w        = (const float*)d_in[11];  // [DEPTH, E, D, F]
    const float* d2b        = (const float*)d_in[12];  // [DEPTH, E, D]
    float*       out        = (float*)d_out;           // [L, V]

    const size_t MF = 1024 * 1024;
    float* ws   = (float*)d_ws;
    float* h    = ws;                 // 2 MF
    float* u    = ws +  2 * MF;       // 2 MF
    float* proj = ws +  4 * MF;       // 3 MF
    float* h1   = ws +  7 * MF;       // 4 MF
    float* h2   = ws + 11 * MF;       // 2 MF
    float* gwts = ws + 13 * MF;       // small
    int*   gidx = (int*)(gwts + 2 * L_);
    unsigned short* ubf  = (unsigned short*)(ws + 14 * MF);  // 2M elems
    unsigned short* obf  = (unsigned short*)(ws + 15 * MF);  // 2M elems
    unsigned short* s1bf = (unsigned short*)(ws + 16 * MF);  // 2M elems
    unsigned short* wbuf = (unsigned short*)(ws + 17 * MF);  // up to 32.77M elems

    dim3 blk(256);

    k_embed<<<L_, blk, 0, stream>>>(x, emb, h);

    for (int i = 0; i < DEPTH; ++i) {
        // ---- attention block ----
        k_rmsnorm<<<L_, blk, 0, stream>>>(h, ln1 + (size_t)i * D_, u, ubf);

        k_cvt_t<<<dim3(D_ / 32, PROJ_ / 32), blk, 0, stream>>>(
            attn_in_w + (size_t)i * D_ * PROJ_, wbuf, D_, PROJ_);
        k_gemm<false><<<dim3(L_ / 128, PROJ_ / 128), blk, 0, stream>>>(
            ubf, wbuf, proj, L_, PROJ_, D_);

        k_rope<<<(L_ * (HQ_ + HKV_) * 32) / 256, blk, 0, stream>>>(proj);
        k_attn<<<(HQ_ * L_) / 8, blk, 0, stream>>>(proj, sink + (size_t)i * HQ_, obf);

        k_cvt_t<<<dim3((HQ_ * HD_) / 32, D_ / 32), blk, 0, stream>>>(
            attn_out_w + (size_t)i * (HQ_ * HD_) * D_, wbuf, HQ_ * HD_, D_);
        k_gemm<true><<<dim3(L_ / 128, D_ / 128), blk, 0, stream>>>(
            obf, wbuf, h, L_, D_, HQ_ * HD_);

        // ---- MoE block (dense over experts, matching the reference) ----
        k_rmsnorm<<<L_, blk, 0, stream>>>(h, ln2 + (size_t)i * D_, u, ubf);
        k_gate<<<L_ / 8, blk, 0, stream>>>(u, gate_w + (size_t)i * D_ * E_, gwts, gidx);

        for (int e = 0; e < E_; ++e) {
            const float* w1 = d1w + ((size_t)i * E_ + e) * 2 * F_ * D_;  // [2F, D]
            const float* b1 = d1b + ((size_t)i * E_ + e) * 2 * F_;
            const float* w2 = d2w + ((size_t)i * E_ + e) * D_ * F_;      // [D, F]
            const float* b2 = d2b + ((size_t)i * E_ + e) * D_;

            k_cvt<<<(2 * F_ * D_) / 1024, blk, 0, stream>>>(w1, wbuf, 2 * F_ * D_);
            k_gemm<false><<<dim3(L_ / 128, (2 * F_) / 128), blk, 0, stream>>>(
                ubf, wbuf, h1, L_, 2 * F_, D_);
            k_swiglu<<<L_, blk, 0, stream>>>(h1, b1, s1bf);
            k_cvt<<<(D_ * F_) / 1024, blk, 0, stream>>>(w2, wbuf, D_ * F_);
            k_gemm<false><<<dim3(L_ / 128, D_ / 128), blk, 0, stream>>>(
                s1bf, wbuf, h2, L_, D_, F_);
            k_moe_accum<<<L_, blk, 0, stream>>>(h, h2, b2, gwts, gidx, e);
        }
    }

    // ---- final norm + tied-embedding logits (bf16 B streams once, L2-resident)
    k_rmsnorm<<<L_, blk, 0, stream>>>(h, lnf, u, ubf);
    k_cvt<<<(V_ * D_) / 1024, blk, 0, stream>>>(emb, wbuf, V_ * D_);
    k_gemm<false><<<dim3(L_ / 128, V_ / 128), blk, 0, stream>>>(
        ubf, wbuf, out, L_, V_, D_);
}
